// LLMValueNetwork_75548474737164
// MI455X (gfx1250) — matrix operand order repaired and checked by reading, not compiled, not run on hardware
//
#include <hip/hip_runtime.h>
#include <hip/hip_bf16.h>
#include <cstdint>

// ---------------------------------------------------------------------------
// MI455X (gfx1250) fused value-network forward.
// ~85 GFLOP vs ~34 MB HBM traffic -> compute bound. Whole network fused per
// group-of-8 batches in one workgroup; intermediates in LDS; matmuls via
// v_wmma_f32_16x16x32_f16 (wave32) with M-blocked register tiling so each
// weight B-fragment is fetched from L2 once per MTB output tiles.
// ---------------------------------------------------------------------------

typedef _Float16 h16v __attribute__((ext_vector_type(16)));
typedef _Float16 h8v  __attribute__((ext_vector_type(8)));
typedef float    f8v  __attribute__((ext_vector_type(8)));

#define G_BATCH   8
#define ROWS      160      // G_BATCH * 20 tokens
#define THREADS   512
#define NWAVES    16

// ----------------------- workspace layout (device global) ------------------
// f16 B-fragments, fragment-ordered: [kchunk][ntile][lane(32)][elem(16)]
//   elem e of lane l maps to B[k][n]:
//     k = kc*32 + (l>=16 ? 16 : 0) + e ,  n = nt*16 + (l&15)
static constexpr size_t OF_W1a = 0;                    // 1kc x 10nt
static constexpr size_t OF_W1b = OF_W1a + 1*10*512;    // 5kc x 8nt
static constexpr size_t OF_W2a = OF_W1b + 5*8*512;     // 4kc x 8nt
static constexpr size_t OF_W2b = OF_W2a + 4*8*512;     // 4kc x 4nt
static constexpr size_t OF_Wa1 = OF_W2b + 4*4*512;     // 4kc x 8nt (top 100 rows)
static constexpr size_t OF_Wa2 = OF_Wa1 + 4*8*512;     // 4kc x 8nt
static constexpr size_t OF_W3a = OF_Wa2 + 4*8*512;     // 2kc x 10nt
static constexpr size_t OF_W3b = OF_W3a + 2*10*512;    // 5kc x 8nt
static constexpr size_t OF_W3c = OF_W3b + 5*8*512;     // 4kc x 8nt
static constexpr size_t FRAG_HALVES = OF_W3c + 4*8*512;     // 130048 halves
static constexpr size_t BIAS_BYTE_OFF = FRAG_HALVES * 2;    // 260096 B
// padded f32 biases (float offsets inside bias region)
static constexpr size_t OB1a = 0;          // 160
static constexpr size_t OB1b = OB1a + 160; // 128
static constexpr size_t OB2a = OB1b + 128; // 128
static constexpr size_t OB2b = OB2a + 128; // 64
static constexpr size_t OBa1 = OB2b + 64;  // 128
static constexpr size_t OBa2 = OBa1 + 128; // 128
static constexpr size_t OB3a = OBa2 + 128; // 160
static constexpr size_t OB3b = OB3a + 160; // 128
static constexpr size_t OB3c = OB3b + 128; // 128

// ----------------------------- prep kernels --------------------------------
__global__ void prep_frag(const float* __restrict__ W, _Float16* __restrict__ frag,
                          int Kdim, int Ndim, int kchunks, int ntiles) {
    int idx = blockIdx.x * blockDim.x + threadIdx.x;
    int total = kchunks * ntiles * 512;
    if (idx >= total) return;
    int e    = idx & 15;
    int l    = (idx >> 4) & 31;
    int tile = idx >> 9;
    int nt   = tile % ntiles;
    int kc   = tile / ntiles;
    int K = kc * 32 + ((l >= 16) ? 16 : 0) + e;
    int N = nt * 16 + (l & 15);
    float v = (K < Kdim && N < Ndim) ? W[(size_t)K * Ndim + N] : 0.0f;
    frag[idx] = (_Float16)v;
}

__global__ void prep_bias(const float* __restrict__ b, float* __restrict__ dst,
                          int n, int npad) {
    int i = blockIdx.x * blockDim.x + threadIdx.x;
    if (i < npad) dst[i] = (i < n) ? b[i] : 0.0f;
}

// ------------------------------ main kernel --------------------------------
struct Params {
    const float* state;
    const _Float16 *fW1a, *fW1b, *fW2a, *fW2b, *fWa1, *fWa2, *fW3a, *fW3b, *fW3c;
    const float    *b1a, *b1b, *b2a, *b2b, *ba1, *ba2, *b3a, *b3b, *b3c;
    const float    *Wa1_full;  // (200,100) row-major; rows 100..199 multiply g
    const float    *Wa3, *ba3, *W3d, *b3d;
    float*          out;
};

__device__ __forceinline__ h16v load_afrag(const _Float16* __restrict__ act,
                                           int actStride, int mt, int kc,
                                           int lg, int lm) {
    // 16-bit A 16x32 layout (contiguous, matches B fragments):
    //   lanes 0-15 : elems 0-15 = K 0..15
    //   lanes 16-31: elems 0-15 = K 16..31
    const _Float16* ar = act + (size_t)(mt * 16 + lm) * actStride + kc * 32 + lg * 16;
    h8v a0 = *(const h8v*)ar;
    h8v a1 = *(const h8v*)(ar + 8);
    return __builtin_shufflevector(a0, a1, 0, 1, 2, 3, 4, 5, 6, 7,
                                           8, 9, 10, 11, 12, 13, 14, 15);
}

// Wave-level tiled GEMM with M-block register blocking:
// each work item = (m-block of MTB tiles, one n-tile); B fragment is loaded
// once per k-chunk and reused by MTB independent WMMA accumulator chains.
template <int MTB>
__device__ __forceinline__ void gemm_lds(
    const _Float16* __restrict__ act, int actStride,
    const _Float16* __restrict__ wfrag,
    const float* __restrict__ bias,            // padded or nullptr
    const float* __restrict__ extraRow,        // LDS f32 [batch][extraStride] or nullptr
    int extraStride, int rowsPerBatch,
    _Float16* __restrict__ outAct, int outStride,
    int mtiles, int ntiles, int kchunks, bool relu,
    int wave, int nwaves, int lane) {
    const int lg = lane >> 4;       // lane group (0: lanes 0-15, 1: lanes 16-31)
    const int lm = lane & 15;
    const int mblocks = mtiles / MTB;
    const int W = mblocks * ntiles;
    for (int w = wave; w < W; w += nwaves) {
        const int mb = w / ntiles;
        const int nt = w % ntiles;
        f8v acc[MTB];
#pragma unroll
        for (int i = 0; i < MTB; ++i) acc[i] = f8v{0.f, 0.f, 0.f, 0.f, 0.f, 0.f, 0.f, 0.f};
        for (int kc = 0; kc < kchunks; ++kc) {
            // B fragment: contiguous 16 halves/lane, fragment-ordered in ws
            const _Float16* bp = wfrag + (((size_t)kc * ntiles + nt) * 32 + lane) * 16;
            h16v b = *(const h16v*)bp;
#pragma unroll
            for (int i = 0; i < MTB; ++i) {
                h16v a = load_afrag(act, actStride, mb * MTB + i, kc, lg, lm);
                acc[i] = __builtin_amdgcn_wmma_f32_16x16x32_f16(false, a, false, b,
                                                                (short)0, acc[i],
                                                                false, false);
            }
        }
        // epilogue: bias (+ per-row extra) + ReLU + cvt f16, write to LDS
        const int n = nt * 16 + lm;
        const float bv = bias ? bias[n] : 0.0f;
#pragma unroll
        for (int i = 0; i < MTB; ++i) {
            const int mt = mb * MTB + i;
#pragma unroll
            for (int r = 0; r < 8; ++r) {
                const int row = mt * 16 + r + lg * 8;  // C layout: VGPR r -> M=r (+8 hi lanes)
                float v = acc[i][r] + bv;
                if (extraRow) v += extraRow[(row / rowsPerBatch) * extraStride + n];
                if (relu) v = v > 0.0f ? v : 0.0f;
                outAct[(size_t)row * outStride + n] = (_Float16)v;
            }
        }
    }
}

__global__ void __launch_bounds__(THREADS, 1)
value_net_fused(Params p) {
    extern __shared__ char smem[];
    _Float16* X   = (_Float16*)(smem + 0);        // [160][32]
    _Float16* S1  = (_Float16*)(smem + 10240);    // [160][160] h1a -> att1
    _Float16* H1  = (_Float16*)(smem + 61440);    // [160][128] h1
    _Float16* S2  = (_Float16*)(smem + 102400);   // [160][128] h2a -> att2
    _Float16* H2  = (_Float16*)(smem + 143360);   // [160][64]  h2
    float* gbuf   = (float*)(smem + 163840);      // [8][100]
    float* gctrb  = (float*)(smem + 167040);      // [8][128]
    float* scores = (float*)(smem + 171136);      // [160]
    float* sexp   = (float*)(smem + 171776);      // [160]
    float* ssum   = (float*)(smem + 172416);      // [8]
    float* self6  = (float*)(smem + 172448);      // [8][6]
    _Float16* JNT = (_Float16*)(smem + 172640);   // [16][64]
    _Float16* M3A = (_Float16*)(smem + 174688);   // [16][160]
    _Float16* M3B = (_Float16*)(smem + 179808);   // [16][128]
    _Float16* M3C = (_Float16*)(smem + 183904);   // [16][128]  (end 188000)

    const int tid  = threadIdx.x;
    const int wave = tid >> 5;
    const int lane = tid & 31;
    const int b0   = blockIdx.x * G_BATCH;
    const float* st = p.state + (size_t)b0 * 20 * 13;

    // ---- stage 0: stage state into LDS (f16, K padded 13->32) -------------
    for (int i = tid; i < ROWS * 32; i += THREADS) {
        int r = i >> 5, c = i & 31;
        float v = (c < 13) ? st[r * 13 + c] : 0.0f;
        X[i] = (_Float16)v;
    }
    for (int i = tid; i < G_BATCH * 6; i += THREADS)
        self6[i] = st[(size_t)(i / 6) * 20 * 13 + (i % 6)];
    for (int i = tid; i < 16 * 64; i += THREADS) JNT[i] = (_Float16)0.0f;
    __syncthreads();

    // ---- mlp1: h1a = relu(X@W1a+b), h1 = relu(h1a@W1b+b) ------------------
    gemm_lds<5>(X, 32,  p.fW1a, p.b1a, nullptr, 0, 1, S1, 160, 10, 10, 1, true, wave, NWAVES, lane);
    __syncthreads();
    gemm_lds<5>(S1, 160, p.fW1b, p.b1b, nullptr, 0, 1, H1, 128, 10, 8, 5, true, wave, NWAVES, lane);
    __syncthreads();

    // ---- g = mean_n(h1);  gctrb = g @ Wa1[100:200] ------------------------
    for (int i = tid; i < G_BATCH * 100; i += THREADS) {
        int bl = i / 100, j = i % 100;
        float s = 0.0f;
        for (int n = 0; n < 20; ++n) s += (float)H1[(size_t)(bl * 20 + n) * 128 + j];
        gbuf[bl * 100 + j] = s * (1.0f / 20.0f);
    }
    __syncthreads();
    for (int i = tid; i < G_BATCH * 128; i += THREADS) {
        int bl = i >> 7, j2 = i & 127;
        float s = 0.0f;
        if (j2 < 100)
            for (int j = 0; j < 100; ++j)
                s += gbuf[bl * 100 + j] * p.Wa1_full[(size_t)(100 + j) * 100 + j2];
        gctrb[i] = s;
    }
    // ---- mlp2: h2a = relu(h1@W2a+b), h2 = h2a@W2b+b (no relu) -------------
    gemm_lds<5>(H1, 128, p.fW2a, p.b2a, nullptr, 0, 1, S2, 128, 10, 8, 4, true, wave, NWAVES, lane);
    __syncthreads();
    gemm_lds<2>(S2, 128, p.fW2b, p.b2b, nullptr, 0, 1, H2, 64, 10, 4, 4, false, wave, NWAVES, lane);
    __syncthreads();

    // ---- attention MLP ----------------------------------------------------
    gemm_lds<5>(H1, 128, p.fWa1, p.ba1, gctrb, 128, 20, S1, 128, 10, 8, 4, true, wave, NWAVES, lane);
    __syncthreads();
    gemm_lds<5>(S1, 128, p.fWa2, p.ba2, nullptr, 0, 1, S2, 128, 10, 8, 4, true, wave, NWAVES, lane);
    __syncthreads();
    if (tid < ROWS) {
        float s = 0.0f;
        for (int j = 0; j < 100; ++j) s += (float)S2[(size_t)tid * 128 + j] * p.Wa3[j];
        s += p.ba3[0];
        scores[tid] = s;
        sexp[tid] = (s != 0.0f) ? expf(s) : 0.0f;   // exp * (score != 0) mask
    }
    __syncthreads();
    if (tid < G_BATCH) {
        float s = 0.0f;
        for (int n = 0; n < 20; ++n) s += sexp[tid * 20 + n];
        ssum[tid] = s;
    }
    __syncthreads();

    // ---- weighted sum + concat(self, weighted) into joint -----------------
    if (tid < G_BATCH * 50) {
        int bl = tid / 50, f = tid % 50;
        float inv = 1.0f / ssum[bl];
        float s = 0.0f;
        for (int n = 0; n < 20; ++n)
            s += sexp[bl * 20 + n] * (float)H2[(size_t)(bl * 20 + n) * 64 + f];
        JNT[bl * 64 + 6 + f] = (_Float16)(s * inv);
    }
    if (tid >= 256 && tid < 256 + G_BATCH * 6) {
        int i = tid - 256;
        JNT[(i / 6) * 64 + (i % 6)] = (_Float16)self6[i];
    }
    __syncthreads();

    // ---- mlp3 (M=16 tile, only rows 0..7 meaningful) ----------------------
    gemm_lds<1>(JNT, 64,  p.fW3a, p.b3a, nullptr, 0, 1, M3A, 160, 1, 10, 2, true, wave, NWAVES, lane);
    __syncthreads();
    gemm_lds<1>(M3A, 160, p.fW3b, p.b3b, nullptr, 0, 1, M3B, 128, 1, 8, 5, true, wave, NWAVES, lane);
    __syncthreads();
    gemm_lds<1>(M3B, 128, p.fW3c, p.b3c, nullptr, 0, 1, M3C, 128, 1, 8, 4, true, wave, NWAVES, lane);
    __syncthreads();
    if (tid < G_BATCH) {
        float s = 0.0f;
        for (int j = 0; j < 100; ++j) s += (float)M3C[(size_t)tid * 128 + j] * p.W3d[j];
        s += p.b3d[0];
        p.out[b0 + tid] = s;
    }
}

// ------------------------------ launcher -----------------------------------
extern "C" void kernel_launch(void* const* d_in, const int* in_sizes, int n_in,
                              void* d_out, int out_size, void* d_ws, size_t ws_size,
                              hipStream_t stream) {
    const float* state = (const float*)d_in[0];
    const float* W1a = (const float*)d_in[1];  const float* B1a = (const float*)d_in[2];
    const float* W1b = (const float*)d_in[3];  const float* B1b = (const float*)d_in[4];
    const float* W2a = (const float*)d_in[5];  const float* B2a = (const float*)d_in[6];
    const float* W2b = (const float*)d_in[7];  const float* B2b = (const float*)d_in[8];
    const float* Wa1 = (const float*)d_in[9];  const float* Ba1 = (const float*)d_in[10];
    const float* Wa2 = (const float*)d_in[11]; const float* Ba2 = (const float*)d_in[12];
    const float* Wa3 = (const float*)d_in[13]; const float* Ba3 = (const float*)d_in[14];
    const float* W3a = (const float*)d_in[15]; const float* B3a = (const float*)d_in[16];
    const float* W3b = (const float*)d_in[17]; const float* B3b = (const float*)d_in[18];
    const float* W3c = (const float*)d_in[19]; const float* B3c = (const float*)d_in[20];
    const float* W3d = (const float*)d_in[21]; const float* B3d = (const float*)d_in[22];

    _Float16* frag = (_Float16*)d_ws;
    float* bws = (float*)((char*)d_ws + BIAS_BYTE_OFF);

    auto lf = [&](const float* W, size_t off, int K, int N, int kc, int nt) {
        int total = kc * nt * 512;
        prep_frag<<<(total + 255) / 256, 256, 0, stream>>>(W, frag + off, K, N, kc, nt);
    };
    lf(W1a, OF_W1a, 13,  150, 1, 10);
    lf(W1b, OF_W1b, 150, 100, 5, 8);
    lf(W2a, OF_W2a, 100, 100, 4, 8);
    lf(W2b, OF_W2b, 100, 50,  4, 4);
    lf(Wa1, OF_Wa1, 100, 100, 4, 8);   // top 100 rows of (200,100)
    lf(Wa2, OF_Wa2, 100, 100, 4, 8);
    lf(W3a, OF_W3a, 56,  150, 2, 10);
    lf(W3b, OF_W3b, 150, 100, 5, 8);
    lf(W3c, OF_W3c, 100, 100, 4, 8);

    auto lb = [&](const float* b, size_t off, int n, int npad) {
        prep_bias<<<(npad + 255) / 256, 256, 0, stream>>>(b, bws + off, n, npad);
    };
    lb(B1a, OB1a, 150, 160); lb(B1b, OB1b, 100, 128);
    lb(B2a, OB2a, 100, 128); lb(B2b, OB2b, 50, 64);
    lb(Ba1, OBa1, 100, 128); lb(Ba2, OBa2, 100, 128);
    lb(B3a, OB3a, 150, 160); lb(B3b, OB3b, 100, 128);
    lb(B3c, OB3c, 100, 128);

    Params p;
    p.state = state;
    p.fW1a = frag + OF_W1a; p.fW1b = frag + OF_W1b;
    p.fW2a = frag + OF_W2a; p.fW2b = frag + OF_W2b;
    p.fWa1 = frag + OF_Wa1; p.fWa2 = frag + OF_Wa2;
    p.fW3a = frag + OF_W3a; p.fW3b = frag + OF_W3b; p.fW3c = frag + OF_W3c;
    p.b1a = bws + OB1a; p.b1b = bws + OB1b;
    p.b2a = bws + OB2a; p.b2b = bws + OB2b;
    p.ba1 = bws + OBa1; p.ba2 = bws + OBa2;
    p.b3a = bws + OB3a; p.b3b = bws + OB3b; p.b3c = bws + OB3c;
    p.Wa1_full = Wa1;
    p.Wa3 = Wa3; p.ba3 = Ba3; p.W3d = W3d; p.b3d = B3d;
    p.out = (float*)d_out;

    int Btot = in_sizes[0] / (20 * 13);          // 32768
    int grid = Btot / G_BATCH;                   // 4096 workgroups
    value_net_fused<<<grid, THREADS, 188000, stream>>>(p);
}
